// KE_H_GNN_82025285419543
// MI455X (gfx1250) — matrix operand order repaired and checked by reading, not compiled, not run on hardware
//
#include <hip/hip_runtime.h>
#include <hip/hip_bf16.h>
#include <stdint.h>

// ---------------------------------------------------------------------------
// Types for CDNA5 WMMA (wave32): D(16x16 f32) = A(16x32 bf16) x B(32x16 bf16) + C
// ---------------------------------------------------------------------------
typedef __attribute__((ext_vector_type(16))) __bf16 v16bf;
typedef __attribute__((ext_vector_type(8)))  float  v8f;

union FragAB { uint4 u[2]; v16bf v; };

__device__ __forceinline__ unsigned short f2bf(float f) {
  unsigned u = __float_as_uint(f);
  u = (u + 0x7FFFu + ((u >> 16) & 1u)) >> 16;   // round-to-nearest-even
  return (unsigned short)u;
}

// Order-preserving float <-> uint mapping for atomicMax-based segment max
__device__ __forceinline__ unsigned encf(float f) {
  unsigned u = __float_as_uint(f);
  return (u & 0x80000000u) ? ~u : (u | 0x80000000u);
}
__device__ __forceinline__ float decf(unsigned u) {
  unsigned b = (u & 0x80000000u) ? (u & 0x7FFFFFFFu) : ~u;
  return __uint_as_float(b);
}

// ---------------------------------------------------------------------------
// Generic bf16 WMMA GEMM:  C[M, NDIM] = A[M, KDIM] * W[KDIM, NDIM]  (+bias/act)
// One wave -> 16x32 output slab (two 16x16 tiles, shared A fragment).
// A: row-major bf16 (activations).  Bp: pre-packed WMMA B-fragment layout.
// EPI: 0 = store f32; 1 = +bias,relu,store bf16; 2 = +bias,store bf16;
//      3 = +bias,store f32.
// ---------------------------------------------------------------------------
template <int KDIM, int NDIM, int EPI>
__global__ __launch_bounds__(128)
void gemm_bf16_wmma(const unsigned short* __restrict__ A,
                    const unsigned short* __restrict__ Bp,
                    const float* __restrict__ bias,
                    void* __restrict__ Cout) {
  constexpr int KT = KDIM / 32;
  const int lane = threadIdx.x & 31;
  const int wave = threadIdx.x >> 5;
  const int M0   = (blockIdx.x * 4 + wave) * 16;   // M divisible by 64
  const int nt0  = blockIdx.y * 2;
  const int r    = lane & 15;
  const int hs   = lane >> 4;

  // A fragment: lane holds row (lane&15); K chunks at hs*8 and hs*8+16
  const unsigned short* Arow = A + (size_t)(M0 + r) * KDIM + hs * 8;
  // Packed B: tile (nt,kt) base = (nt*KT + kt)*512 elements, lane*16 within
  const unsigned short* B0 = Bp + ((size_t)(nt0    ) * KT) * 512 + lane * 16;
  const unsigned short* B1 = Bp + ((size_t)(nt0 + 1) * KT) * 512 + lane * 16;

  v8f acc0 = {}; v8f acc1 = {};
#pragma unroll
  for (int kt = 0; kt < KT; ++kt) {
    FragAB fa, fb0, fb1;
    const uint4* pa = reinterpret_cast<const uint4*>(Arow + kt * 32);
    fa.u[0] = pa[0];                 // K = hs*8 .. +7
    fa.u[1] = pa[2];                 // K = hs*8+16 .. +23
    const uint4* pb0 = reinterpret_cast<const uint4*>(B0 + (size_t)kt * 512);
    fb0.u[0] = pb0[0]; fb0.u[1] = pb0[1];
    const uint4* pb1 = reinterpret_cast<const uint4*>(B1 + (size_t)kt * 512);
    fb1.u[0] = pb1[0]; fb1.u[1] = pb1[1];
    acc0 = __builtin_amdgcn_wmma_f32_16x16x32_bf16(false, fa.v, false, fb0.v,
                                                   (short)0, acc0, false, false);
    acc1 = __builtin_amdgcn_wmma_f32_16x16x32_bf16(false, fa.v, false, fb1.v,
                                                   (short)0, acc1, false, false);
  }

  // C layout: lane -> col = N0 + (lane&15); vgpr i -> row = M0 + hs*8 + i
  const int colA = nt0 * 16 + r;
  const int colB = colA + 16;
  const int rowB = M0 + hs * 8;

  if constexpr (EPI == 0) {
    float* C = (float*)Cout;
#pragma unroll
    for (int i = 0; i < 8; ++i) {
      C[(size_t)(rowB + i) * NDIM + colA] = acc0[i];
      C[(size_t)(rowB + i) * NDIM + colB] = acc1[i];
    }
  } else if constexpr (EPI == 3) {
    float* C = (float*)Cout;
    const float ba = bias[colA], bb = bias[colB];
#pragma unroll
    for (int i = 0; i < 8; ++i) {
      C[(size_t)(rowB + i) * NDIM + colA] = acc0[i] + ba;
      C[(size_t)(rowB + i) * NDIM + colB] = acc1[i] + bb;
    }
  } else {
    unsigned short* C = (unsigned short*)Cout;
    const float ba = bias[colA], bb = bias[colB];
#pragma unroll
    for (int i = 0; i < 8; ++i) {
      float v0 = acc0[i] + ba;
      float v1 = acc1[i] + bb;
      if constexpr (EPI == 1) { v0 = fmaxf(v0, 0.f); v1 = fmaxf(v1, 0.f); }
      C[(size_t)(rowB + i) * NDIM + colA] = f2bf(v0);
      C[(size_t)(rowB + i) * NDIM + colB] = f2bf(v1);
    }
  }
}

// ---------------------------------------------------------------------------
// Pack f32 weight [K,N] row-major -> bf16 WMMA B-fragment tiles.
// Tile (kt,nt): element idx = lane*16 + e holds W[k][n] with
//   n = nt*16 + (lane&15),  k = kt*32 + (e>>3)*16 + (lane>>4)*8 + (e&7)
// ---------------------------------------------------------------------------
__global__ void pack_weight_bf16(const float* __restrict__ W,
                                 unsigned short* __restrict__ dst,
                                 int K, int N) {
  const int idx = blockIdx.x * blockDim.x + threadIdx.x;
  const int KT = K / 32;
  const int total = KT * (N / 16) * 512;
  if (idx >= total) return;
  const int tile = idx >> 9;
  const int w    = idx & 511;
  const int lane = w >> 4;
  const int e    = w & 15;
  const int kt = tile % KT, nt = tile / KT;
  const int n = nt * 16 + (lane & 15);
  const int k = kt * 32 + (e >> 3) * 16 + (lane >> 4) * 8 + (e & 7);   // table K order
  dst[idx] = f2bf(W[(size_t)k * N + n]);
}

// f32 -> bf16 elementwise (float4 granularity)
__global__ void cvt_f32_bf16(const float* __restrict__ src,
                             unsigned short* __restrict__ dst, int n4) {
  const int i = blockIdx.x * blockDim.x + threadIdx.x;
  if (i >= n4) return;
  const float4 f = reinterpret_cast<const float4*>(src)[i];
  ushort4 o;
  o.x = f2bf(f.x); o.y = f2bf(f.y); o.z = f2bf(f.z); o.w = f2bf(f.w);
  reinterpret_cast<ushort4*>(dst)[i] = o;
}

// Per-node attention scores: a_s[m] = h[m,:].att_src, a_d[m] = h[m,:].att_dst
__global__ void node_scores(const float* __restrict__ h,
                            const float* __restrict__ att_s,
                            const float* __restrict__ att_d,
                            float* __restrict__ as_, float* __restrict__ ad_,
                            int M) {
  const int m = blockIdx.x * blockDim.x + threadIdx.x;
  if (m >= M) return;
  const float4* hp = reinterpret_cast<const float4*>(h + (size_t)m * 128);
  const float4* s4 = reinterpret_cast<const float4*>(att_s);
  const float4* d4 = reinterpret_cast<const float4*>(att_d);
  float ss = 0.f, dd = 0.f;
#pragma unroll 8
  for (int i = 0; i < 32; ++i) {
    const float4 hv = hp[i], sv = s4[i], dv = d4[i];
    ss += hv.x * sv.x + hv.y * sv.y + hv.z * sv.z + hv.w * sv.w;
    dd += hv.x * dv.x + hv.y * dv.y + hv.z * dv.z + hv.w * dv.w;
  }
  as_[m] = ss; ad_[m] = dd;
}

__device__ __forceinline__ void edge_pair(const long long* ei, int e, int E,
                                          int& src, int& dst) {
  if (e < E) { src = (int)ei[e]; dst = (int)ei[(size_t)E + e]; }
  else       { src = dst = e - E; }            // self loops appended
}

// Pass 1: e = leakyrelu(a_s[src]+a_d[dst]); segment max over dst (atomicMax)
__global__ void edge_max(const long long* __restrict__ ei,
                         const float* __restrict__ as_,
                         const float* __restrict__ ad_,
                         float* __restrict__ ebuf,
                         unsigned* __restrict__ segm,
                         int E, int Etot, int N, int total) {
  const int idx = blockIdx.x * blockDim.x + threadIdx.x;
  if (idx >= total) return;
  const int b = idx / Etot;
  const int e = idx - b * Etot;
  int src, dst; edge_pair(ei, e, E, src, dst);
  float v = as_[b * N + src] + ad_[b * N + dst];
  v = v > 0.f ? v : 0.2f * v;                  // negative_slope = 0.2
  ebuf[idx] = v;
  atomicMax(&segm[b * N + dst], encf(v));
}

// Pass 2: ex = exp(e - max[dst]); segment sum over dst
__global__ void edge_sum(const long long* __restrict__ ei,
                         float* __restrict__ ebuf,
                         const unsigned* __restrict__ segm,
                         float* __restrict__ den,
                         int E, int Etot, int N, int total) {
  const int idx = blockIdx.x * blockDim.x + threadIdx.x;
  if (idx >= total) return;
  const int b = idx / Etot;
  const int e = idx - b * Etot;
  int src, dst; edge_pair(ei, e, E, src, dst);
  const float m  = decf(segm[b * N + dst]);
  const float ex = __expf(ebuf[idx] - m);
  ebuf[idx] = ex;
  unsafeAtomicAdd(&den[b * N + dst], ex);
}

// Pass 3: agg[dst,:] += (ex/den[dst]) * h[src,:]   (one wave per edge)
__global__ void edge_agg(const long long* __restrict__ ei,
                         const float* __restrict__ ebuf,
                         const float* __restrict__ den,
                         const float* __restrict__ h,
                         float* __restrict__ agg,
                         int E, int Etot, int N, int total) {
  const int warp = (blockIdx.x * blockDim.x + threadIdx.x) >> 5;
  const int lane = threadIdx.x & 31;
  if (warp >= total) return;
  const int b = warp / Etot;
  const int e = warp - b * Etot;
  int src, dst; edge_pair(ei, e, E, src, dst);
  const float alpha = ebuf[warp] / den[b * N + dst];
  const float4 hv =
      reinterpret_cast<const float4*>(h + (size_t)(b * N + src) * 128)[lane];
  float* ap = agg + (size_t)(b * N + dst) * 128 + lane * 4;
  unsafeAtomicAdd(ap + 0, alpha * hv.x);
  unsafeAtomicAdd(ap + 1, alpha * hv.y);
  unsafeAtomicAdd(ap + 2, alpha * hv.z);
  unsafeAtomicAdd(ap + 3, alpha * hv.w);
}

// hybrid[m, 0:128] = bf16(relu(agg + bias_gat)); hybrid[m, 128:192] = bf16(ext)
__global__ void build_hybrid(const float* __restrict__ agg,
                             const float* __restrict__ bgat,
                             const float* __restrict__ ext,
                             unsigned short* __restrict__ hyb, int M) {
  const int idx = blockIdx.x * blockDim.x + threadIdx.x;
  if (idx >= M * 192) return;
  const int m = idx / 192;
  const int c = idx - m * 192;
  float v;
  if (c < 128) {
    v = fmaxf(agg[(size_t)m * 128 + c] + bgat[c], 0.f);
  } else {
    v = ext[(size_t)m * 64 + (c - 128)];
  }
  hyb[idx] = f2bf(v);
}

// ---------------------------------------------------------------------------
extern "C" void kernel_launch(void* const* d_in, const int* in_sizes, int n_in,
                              void* d_out, int out_size, void* d_ws,
                              size_t ws_size, hipStream_t stream) {
  (void)in_sizes; (void)n_in; (void)out_size; (void)ws_size;
  constexpr int Bc = 4, Nn = 50000, Ee = 800000;
  constexpr int DIN = 128, H = 128, EXT = 64, MLR = 256, OD = 64;
  constexpr int M = Bc * Nn;        // 200000 (divisible by 64)
  constexpr int Etot = Ee + Nn;     // 850000 (with self loops)

  const float*     x    = (const float*)d_in[0];
  const long long* ei   = (const long long*)d_in[1];   // int64 per reference
  const float*     ext  = (const float*)d_in[2];
  const float*     W    = (const float*)d_in[3];
  const float*     atts = (const float*)d_in[4];
  const float*     attd = (const float*)d_in[5];
  const float*     bgat = (const float*)d_in[6];
  const float*     W1   = (const float*)d_in[7];
  const float*     b1   = (const float*)d_in[8];
  const float*     W2   = (const float*)d_in[9];
  const float*     b2   = (const float*)d_in[10];
  const float*     Wo   = (const float*)d_in[11];
  const float*     bo   = (const float*)d_in[12];

  char* ws = (char*)d_ws;
  size_t off = 0;
  auto alloc = [&](size_t bytes) -> void* {
    void* p = ws + off;
    off += (bytes + 255) & ~size_t(255);
    return p;
  };
  unsigned short* xbf  = (unsigned short*)alloc((size_t)M * DIN * 2);
  float*          h    = (float*)alloc((size_t)M * H * 4);
  float*          agg  = (float*)alloc((size_t)M * H * 4);
  unsigned short* hyb  = (unsigned short*)alloc((size_t)M * (H + EXT) * 2);
  float*          ebuf = (float*)alloc((size_t)Bc * Etot * 4);
  float*          asb  = (float*)alloc((size_t)M * 4);
  float*          adb  = (float*)alloc((size_t)M * 4);
  unsigned*       segm = (unsigned*)alloc((size_t)M * 4);
  float*          den  = (float*)alloc((size_t)M * 4);
  unsigned short* pW   = (unsigned short*)alloc((size_t)DIN * H * 2);
  unsigned short* pW1  = (unsigned short*)alloc((size_t)(H + EXT) * MLR * 2);
  unsigned short* pW2  = (unsigned short*)alloc((size_t)MLR * H * 2);
  unsigned short* pWo  = (unsigned short*)alloc((size_t)H * OD * 2);
  // buffer reuse (stream ordered): z over h, feat over xbf
  unsigned short* zbf   = (unsigned short*)h;
  unsigned short* featb = (unsigned short*)xbf;

  hipMemsetAsync(agg,  0, (size_t)M * H * 4, stream);
  hipMemsetAsync(segm, 0, (size_t)M * 4,     stream);   // enc-min
  hipMemsetAsync(den,  0, (size_t)M * 4,     stream);

  auto pack = [&](const float* s, unsigned short* d, int K, int Nc) {
    const int total = (K / 32) * (Nc / 16) * 512;
    pack_weight_bf16<<<(total + 255) / 256, 256, 0, stream>>>(s, d, K, Nc);
  };
  pack(W,  pW,  DIN,     H);
  pack(W1, pW1, H + EXT, MLR);
  pack(W2, pW2, MLR,     H);
  pack(Wo, pWo, H,       OD);

  cvt_f32_bf16<<<(M * DIN / 4 + 255) / 256, 256, 0, stream>>>(x, xbf,
                                                              M * DIN / 4);

  const dim3 blk(128);
  // GEMM1: h = x @ W   (f32 out for attention / message gather)
  gemm_bf16_wmma<DIN, H, 0>
      <<<dim3(M / 64, H / 32), blk, 0, stream>>>(xbf, pW, nullptr, h);

  node_scores<<<(M + 255) / 256, 256, 0, stream>>>(h, atts, attd, asb, adb, M);

  const int totE = Bc * Etot;
  edge_max<<<(totE + 255) / 256, 256, 0, stream>>>(ei, asb, adb, ebuf, segm,
                                                   Ee, Etot, Nn, totE);
  edge_sum<<<(totE + 255) / 256, 256, 0, stream>>>(ei, ebuf, segm, den,
                                                   Ee, Etot, Nn, totE);
  edge_agg<<<((size_t)totE * 32 + 255) / 256, 256, 0, stream>>>(
      ei, ebuf, den, h, agg, Ee, Etot, Nn, totE);

  build_hybrid<<<(M * 192 + 255) / 256, 256, 0, stream>>>(agg, bgat, ext, hyb,
                                                          M);

  // GEMM2: z = relu(hybrid @ W1 + b1)  -> bf16
  gemm_bf16_wmma<H + EXT, MLR, 1>
      <<<dim3(M / 64, MLR / 32), blk, 0, stream>>>(hyb, pW1, b1, zbf);
  // GEMM3: feat = z @ W2 + b2          -> bf16
  gemm_bf16_wmma<MLR, H, 2>
      <<<dim3(M / 64, H / 32), blk, 0, stream>>>(zbf, pW2, b2, featb);
  // GEMM4: out = feat @ W_out + b_out  -> f32 to d_out
  gemm_bf16_wmma<H, OD, 3>
      <<<dim3(M / 64, OD / 32), blk, 0, stream>>>(featb, pWo, bo,
                                                  (float*)d_out);
}
